// SparseAttentionAdapter_39719857553506
// MI455X (gfx1250) — matrix operand order repaired and checked by reading, not compiled, not run on hardware
//
#include <hip/hip_runtime.h>
#include <hip/hip_bf16.h>

// ---------------------------------------------------------------------------
// Problem constants (B=1)
#define Hh   16
#define HKVh 4
#define Gg   4
#define Nn   2048
#define Dd   64
#define CBc  16
#define Ww   128          // N / CB blocks
#define QT   128          // N / 16 query tiles
#define HIDh 1024
#define NEGF (-1.0e10f)

#define CKROWS 144        // mem_k + 128 blocks, padded to 9 key tiles
#define CVS    160        // cv^T key stride (5 chunks of 32)

typedef __attribute__((ext_vector_type(16))) __bf16 v16bf;
typedef __attribute__((ext_vector_type(8)))  __bf16 v8bf;
typedef __attribute__((ext_vector_type(8)))  float  v8f;

__device__ __forceinline__ v16bf cat16(v8bf lo, v8bf hi) {
  return __builtin_shufflevector(lo, hi, 0, 1, 2, 3, 4, 5, 6, 7,
                                 8, 9, 10, 11, 12, 13, 14, 15);
}
// A fragment (16x32 bf16): row pointer + K base; lane half selects K offset.
// K(e) = half*16 + e  -> one contiguous 16-element run.
__device__ __forceinline__ v16bf loadA(const __bf16* rowp, int kbase, int half) {
  return *(const v16bf*)(rowp + kbase + half * 16);
}
// B fragment (32x16 bf16): column pointer + K base; K(e) = half*16 + e
// -> one contiguous 16-element run (32 bytes).
__device__ __forceinline__ v16bf loadB(const __bf16* colp, int kbase, int half) {
  return *(const v16bf*)(colp + kbase + half * 16);
}

__device__ __forceinline__ v8f wmma_bf16(v16bf a, v16bf b, v8f c) {
  return __builtin_amdgcn_wmma_f32_16x16x32_bf16(false, a, false, b,
                                                 (short)0, c, false, false);
}

// ---------------------------------------------------------------------------
// 0a) f32 -> bf16 elementwise convert
__global__ void cvt_bf16_kernel(const float* __restrict__ in,
                                __bf16* __restrict__ out, int n) {
  const int i = blockIdx.x * 256 + threadIdx.x;
  if (i < n) out[i] = (__bf16)in[i];
}

// 0b) v (HKV,N,D) f32 -> vT (HKV,D,N) bf16
__global__ void transpose_v_kernel(const float* __restrict__ v,
                                   __bf16* __restrict__ vT) {
  const int i = blockIdx.x * 256 + threadIdx.x;
  if (i >= HKVh * Nn * Dd) return;
  const int kh = i / (Nn * Dd), r = i % (Nn * Dd);
  const int pos = r / Dd, d = r % Dd;
  vT[(kh * Dd + d) * Nn + pos] = (__bf16)v[i];
}

// 0c) fill mem_k row (key 0) of ckb and key 0 column of cvT
__global__ void memkv_fill_kernel(const float* __restrict__ memkv,
                                  __bf16* __restrict__ ckb,
                                  __bf16* __restrict__ cvT) {
  const int kh = blockIdx.x, d = threadIdx.x;
  ckb[(kh * CKROWS + 0) * Dd + d] = (__bf16)memkv[kh * Dd + d];
  cvT[(kh * Dd + d) * CVS + 0] = (__bf16)memkv[(HKVh + kh) * Dd + d];
}

// ---------------------------------------------------------------------------
// 1) Block compression: ck/cv = rmsnorm(mean of 16 rows); bf16 outputs.
//    ckb: [kh][key=1+wb][d] row-major; cvT: [kh][d][key=1+wb] key-major.
__global__ void compress_kernel(const float* __restrict__ k,
                                const float* __restrict__ v,
                                const float* __restrict__ kscale,
                                const float* __restrict__ vscale,
                                __bf16* __restrict__ ckb,
                                __bf16* __restrict__ cvT) {
  const int bid = blockIdx.x;            // HKV * W blocks
  const int kh = bid / Ww, wb = bid % Ww;
  const int t = threadIdx.x;             // 0..63 = d index
  float mk = 0.f, mv = 0.f;
  const int base = (kh * Nn + wb * CBc) * Dd + t;
  for (int r = 0; r < CBc; ++r) {
    mk += k[base + r * Dd];
    mv += v[base + r * Dd];
  }
  mk *= (1.f / CBc);
  mv *= (1.f / CBc);
  __shared__ float sk[Dd], sv[Dd];
  sk[t] = mk * mk; sv[t] = mv * mv;
  __syncthreads();
  float ssk = 0.f, ssv = 0.f;
  for (int i = 0; i < Dd; ++i) { ssk += sk[i]; ssv += sv[i]; }
  const float rk = rsqrtf(ssk * (1.f / Dd) + 1e-6f);
  const float rv = rsqrtf(ssv * (1.f / Dd) + 1e-6f);
  ckb[(kh * CKROWS + 1 + wb) * Dd + t] = (__bf16)(mk * rk * kscale[t]);
  cvT[(kh * Dd + t) * CVS + 1 + wb] = (__bf16)(mv * rv * vscale[t]);
}

// ---------------------------------------------------------------------------
// 2) Compressed attention + importance accumulation
#define NKT 9     // 9 key tiles of 16 => 144 cols (129 valid)
#define NKP 160
__global__ void cmp_attn_kernel(const float* __restrict__ q,
                                const __bf16* __restrict__ ckb,
                                const __bf16* __restrict__ cvT,
                                float* __restrict__ out_c,
                                float* __restrict__ imp) {
  const int bid = blockIdx.x;                // H * QT blocks, 32 threads
  const int h = bid / QT, qt = bid % QT;
  const int kh = h / Gg;
  const int lane = threadIdx.x;
  const int nl = lane & 15, half = lane >> 4;
  const float scale = 0.125f;                // D^-0.5

  __shared__ __align__(32) __bf16 Qsb[16 * Dd];
  __shared__ float S[16 * NKP];
  __shared__ __align__(32) __bf16 Pb[16 * NKP];

  for (int i = lane; i < 16 * Dd; i += 32) {
    const int m = i / Dd, d = i % Dd;
    Qsb[i] = (__bf16)q[(h * Nn + qt * 16 + m) * Dd + d];
  }
  __syncthreads();

  const v16bf a0 = loadA(&Qsb[nl * Dd], 0, half);
  const v16bf a1 = loadA(&Qsb[nl * Dd], 32, half);

  for (int j = 0; j < NKT; ++j) {
    const int key = j * 16 + nl;            // this lane's B column (key index)
    const __bf16* kp = &ckb[(kh * CKROWS + key) * Dd];
    v8f c = {0.f, 0.f, 0.f, 0.f, 0.f, 0.f, 0.f, 0.f};
    c = wmma_bf16(a0, loadB(kp, 0, half), c);
    c = wmma_bf16(a1, loadB(kp, 32, half), c);
#pragma unroll
    for (int r = 0; r < 8; ++r) {
      const int m = r + half * 8;
      const int qpos = qt * 16 + m;
      const bool valid = (key == 0) || (key <= Ww && (key * CBc - 1) < qpos);
      const float s = valid ? c[r] * scale : NEGF;
      S[m * NKP + j * 16 + nl] = s;
      if (key >= 1 && key <= Ww)
        atomicAdd(&imp[(kh * Nn + qpos) * Ww + (key - 1)], s * (1.f / Gg));
    }
  }
  __syncthreads();

  if (lane < 16) {                           // fp32 softmax, one row per lane
    float mx = NEGF;
    for (int c = 0; c < NKT * 16; ++c) mx = fmaxf(mx, S[lane * NKP + c]);
    float sum = 0.f;
    for (int c = 0; c < NKT * 16; ++c) {
      const float e = __expf(S[lane * NKP + c] - mx);
      S[lane * NKP + c] = e; sum += e;
    }
    const float inv = 1.f / sum;
    for (int c = 0; c < NKT * 16; ++c)
      Pb[lane * NKP + c] = (__bf16)(S[lane * NKP + c] * inv);
    for (int c = NKT * 16; c < NKP; ++c) Pb[lane * NKP + c] = (__bf16)0.f;
  }
  __syncthreads();

  for (int dt = 0; dt < 4; ++dt) {           // P(16x160) x CV(160x16)
    const __bf16* vp = &cvT[(kh * Dd + dt * 16 + nl) * CVS];
    v8f o = {0.f, 0.f, 0.f, 0.f, 0.f, 0.f, 0.f, 0.f};
    for (int kc = 0; kc < 5; ++kc) {
      const v16bf a = loadA(&Pb[nl * NKP], kc * 32, half);
      o = wmma_bf16(a, loadB(vp, kc * 32, half), o);
    }
#pragma unroll
    for (int r = 0; r < 8; ++r) {
      const int m = r + half * 8;
      out_c[(h * Nn + qt * 16 + m) * Dd + dt * 16 + nl] = o[r];
    }
  }
}

// ---------------------------------------------------------------------------
// 3) Top-4 block selection (strict '>' == jax top_k lowest-index ties;
//    softmax is monotone so group-averaged masked logits suffice).
__global__ void topk_kernel(const float* __restrict__ imp, int* __restrict__ sel) {
  const int tid = blockIdx.x * blockDim.x + threadIdx.x;   // HKV*N rows
  if (tid >= HKVh * Nn) return;
  const float* row = &imp[tid * Ww];
  int chosen[4];
  for (int s = 0; s < 4; ++s) {
    float best = -3.4e38f; int bi = 0;
    for (int i = 0; i < Ww; ++i) {
      bool used = false;
      for (int p = 0; p < s; ++p) used |= (chosen[p] == i);
      const float val = row[i];
      if (!used && val > best) { best = val; bi = i; }
    }
    chosen[s] = bi;
    sel[tid * 4 + s] = bi;
  }
}

// ---------------------------------------------------------------------------
// 4) Fine attention over 4 selected blocks (gathered, VALU; tiny FLOP share)
__global__ void fine_attn_kernel(const float* __restrict__ q,
                                 const float* __restrict__ k,
                                 const float* __restrict__ v,
                                 const int* __restrict__ sel,
                                 float* __restrict__ out_f) {
  const int bid = blockIdx.x;                // H*N blocks, 64 threads
  const int h = bid / Nn, row = bid % Nn;
  const int kh = h / Gg;
  const int t = threadIdx.x;
  __shared__ float qs[Dd], sv[64], ps[64];
  __shared__ int bsel[4];
  qs[t] = q[(h * Nn + row) * Dd + t];
  if (t < 4) bsel[t] = sel[(kh * Nn + row) * 4 + t];
  __syncthreads();
  const int kp = bsel[t >> 4] * CBc + (t & 15);
  const float* kr = &k[(kh * Nn + kp) * Dd];
  float dot = 0.f;
  for (int d = 0; d < Dd; ++d) dot += qs[d] * kr[d];
  sv[t] = (kp <= row) ? dot * 0.125f : NEGF;
  __syncthreads();
  float mx = NEGF;
  for (int i = 0; i < 64; ++i) mx = fmaxf(mx, sv[i]);
  float sum = 0.f;
  for (int i = 0; i < 64; ++i) sum += __expf(sv[i] - mx);
  ps[t] = __expf(sv[t] - mx) / sum;
  __syncthreads();
  float acc = 0.f;
  for (int kk = 0; kk < 64; ++kk) {
    const int kp2 = bsel[kk >> 4] * CBc + (kk & 15);
    acc += ps[kk] * v[(kh * Nn + kp2) * Dd + t];
  }
  out_f[(h * Nn + row) * Dd + t] = acc;
}

// ---------------------------------------------------------------------------
// 5) Sliding-window attention (WIN=64): 5 key tiles (80 keys) per q tile
#define WKT 5
#define WKP 96
__global__ void win_attn_kernel(const float* __restrict__ q,
                                const __bf16* __restrict__ kb,
                                const __bf16* __restrict__ vT,
                                float* __restrict__ out_s) {
  const int bid = blockIdx.x;                // H * QT blocks, 32 threads
  const int h = bid / QT, qt = bid % QT;
  const int kh = h / Gg;
  const int lane = threadIdx.x;
  const int nl = lane & 15, half = lane >> 4;
  const float scale = 0.125f;

  __shared__ __align__(32) __bf16 Qsb[16 * Dd];
  __shared__ float S[16 * WKP];
  __shared__ __align__(32) __bf16 Pb[16 * WKP];

  for (int i = lane; i < 16 * Dd; i += 32) {
    const int m = i / Dd, d = i % Dd;
    Qsb[i] = (__bf16)q[(h * Nn + qt * 16 + m) * Dd + d];
  }
  __syncthreads();

  const v16bf a0 = loadA(&Qsb[nl * Dd], 0, half);
  const v16bf a1 = loadA(&Qsb[nl * Dd], 32, half);

  for (int jj = 0; jj < WKT; ++jj) {
    const int jt = qt - 4 + jj;
    if (jt < 0) {
#pragma unroll
      for (int r = 0; r < 8; ++r) S[(r + half * 8) * WKP + jj * 16 + nl] = NEGF;
      continue;
    }
    const int key = jt * 16 + nl;
    const __bf16* kp = &kb[(kh * Nn + key) * Dd];
    v8f c = {0.f, 0.f, 0.f, 0.f, 0.f, 0.f, 0.f, 0.f};
    c = wmma_bf16(a0, loadB(kp, 0, half), c);
    c = wmma_bf16(a1, loadB(kp, 32, half), c);
#pragma unroll
    for (int r = 0; r < 8; ++r) {
      const int m = r + half * 8;
      const int qpos = qt * 16 + m;
      const bool valid = (key <= qpos) && (key > qpos - 64);
      S[m * WKP + jj * 16 + nl] = valid ? c[r] * scale : NEGF;
    }
  }
  __syncthreads();

  if (lane < 16) {
    float mx = NEGF;
    for (int c = 0; c < WKT * 16; ++c) mx = fmaxf(mx, S[lane * WKP + c]);
    float sum = 0.f;
    for (int c = 0; c < WKT * 16; ++c) {
      const float e = __expf(S[lane * WKP + c] - mx);
      S[lane * WKP + c] = e; sum += e;
    }
    const float inv = 1.f / sum;
    for (int c = 0; c < WKT * 16; ++c)
      Pb[lane * WKP + c] = (__bf16)(S[lane * WKP + c] * inv);
    for (int c = WKT * 16; c < WKP; ++c) Pb[lane * WKP + c] = (__bf16)0.f;
  }
  __syncthreads();

  for (int dt = 0; dt < 4; ++dt) {
    const __bf16* vrow = &vT[(kh * Dd + dt * 16 + nl) * Nn];
    v8f o = {0.f, 0.f, 0.f, 0.f, 0.f, 0.f, 0.f, 0.f};
    for (int kc = 0; kc < 3; ++kc) {
      const v16bf a = loadA(&Pb[nl * WKP], kc * 32, half);
      const int base = (qt - 4) * 16 + kc * 32 + half * 16;   // abs key of elem 0
      v16bf b;
      if (base >= 0 && base + 16 <= Nn) {
        b = *(const v16bf*)(vrow + base);
      } else {
#pragma unroll
        for (int e = 0; e < 16; ++e) {
          const int abskey = base + e;
          b[e] = (abskey >= 0 && abskey < Nn) ? vrow[abskey] : (__bf16)0.f;
        }
      }
      o = wmma_bf16(a, b, o);
    }
#pragma unroll
    for (int r = 0; r < 8; ++r) {
      const int m = r + half * 8;
      out_s[(h * Nn + qt * 16 + m) * Dd + dt * 16 + nl] = o[r];
    }
  }
}

// ---------------------------------------------------------------------------
// 6) Gate GEMM: sigmoid(hidden(2048x1024) @ combiner_w^T(1024x48) + b)
//    One wave per 16-row tile; A reused across the 3 column tiles.
__global__ void gates_kernel(const __bf16* __restrict__ hb,
                             const __bf16* __restrict__ cwb,
                             const float* __restrict__ cb,
                             float* __restrict__ gates) {
  const int mt = blockIdx.x;                 // N/16 blocks, 32 threads
  const int lane = threadIdx.x;
  const int nl = lane & 15, half = lane >> 4;
  const __bf16* arow = &hb[(mt * 16 + nl) * HIDh];
  const __bf16* b0p = &cwb[(0 + nl) * HIDh];
  const __bf16* b1p = &cwb[(16 + nl) * HIDh];
  const __bf16* b2p = &cwb[(32 + nl) * HIDh];
  v8f o0 = {0.f, 0.f, 0.f, 0.f, 0.f, 0.f, 0.f, 0.f};
  v8f o1 = o0, o2 = o0;
  for (int kc = 0; kc < HIDh / 32; ++kc) {
    if (kc + 1 < HIDh / 32) {                // prefetch next K-chunk
      __builtin_prefetch(arow + (kc + 1) * 32, 0, 1);
      __builtin_prefetch(b0p + (kc + 1) * 32, 0, 1);
    }
    const v16bf a = loadA(arow, kc * 32, half);
    o0 = wmma_bf16(a, loadB(b0p, kc * 32, half), o0);
    o1 = wmma_bf16(a, loadB(b1p, kc * 32, half), o1);
    o2 = wmma_bf16(a, loadB(b2p, kc * 32, half), o2);
  }
#pragma unroll
  for (int r = 0; r < 8; ++r) {
    const int m = r + half * 8;
    const int row = mt * 16 + m;
    float x0 = o0[r] + cb[nl];
    float x1 = o1[r] + cb[16 + nl];
    float x2 = o2[r] + cb[32 + nl];
    gates[row * (3 * Hh) + nl] = 1.f / (1.f + __expf(-x0));
    gates[row * (3 * Hh) + 16 + nl] = 1.f / (1.f + __expf(-x1));
    gates[row * (3 * Hh) + 32 + nl] = 1.f / (1.f + __expf(-x2));
  }
}

// ---------------------------------------------------------------------------
// 7) Elementwise gate combine into bf16 (N, HID)
__global__ void combine_kernel(const float* __restrict__ gates,
                               const float* __restrict__ oc,
                               const float* __restrict__ of_,
                               const float* __restrict__ os_,
                               __bf16* __restrict__ comb) {
  const int i = blockIdx.x * 256 + threadIdx.x;
  if (i >= Nn * HIDh) return;
  const int n = i / HIDh, c = i % HIDh;
  const int h = c / Dd, d = c % Dd;
  const float g0 = gates[n * (3 * Hh) + h * 3 + 0];
  const float g1 = gates[n * (3 * Hh) + h * 3 + 1];
  const float g2 = gates[n * (3 * Hh) + h * 3 + 2];
  const int o = (h * Nn + n) * Dd + d;
  comb[i] = (__bf16)(g0 * oc[o] + g1 * of_[o] + g2 * os_[o]);
}

// ---------------------------------------------------------------------------
// 8) Final projection: out = comb(2048x1024) @ combine_w^T(1024x1024)
//    8 column tiles per wave: each A fragment feeds 8 WMMAs; next-chunk
//    prefetch overlaps L2 latency with the WMMA issue stream.
#define FGT 8     // column tiles per wave
__global__ void final_gemm_kernel(const __bf16* __restrict__ comb,
                                  const __bf16* __restrict__ wb,
                                  float* __restrict__ out) {
  const int bid = blockIdx.x;                // (N/16)*(HID/(16*FGT)), 32 threads
  const int mt = bid / (HIDh / (16 * FGT)), og = bid % (HIDh / (16 * FGT));
  const int lane = threadIdx.x;
  const int nl = lane & 15, half = lane >> 4;
  const __bf16* arow = &comb[(mt * 16 + nl) * HIDh];
  const __bf16* bp[FGT];
#pragma unroll
  for (int t = 0; t < FGT; ++t) bp[t] = &wb[((og * FGT + t) * 16 + nl) * HIDh];
  v8f acc[FGT];
#pragma unroll
  for (int t = 0; t < FGT; ++t)
    acc[t] = (v8f){0.f, 0.f, 0.f, 0.f, 0.f, 0.f, 0.f, 0.f};
  for (int kc = 0; kc < HIDh / 32; ++kc) {
    if (kc + 1 < HIDh / 32) {                // prefetch next K-chunk
      __builtin_prefetch(arow + (kc + 1) * 32, 0, 1);
      __builtin_prefetch(bp[0] + (kc + 1) * 32, 0, 1);
      __builtin_prefetch(bp[4] + (kc + 1) * 32, 0, 1);
    }
    const v16bf a = loadA(arow, kc * 32, half);
#pragma unroll
    for (int t = 0; t < FGT; ++t)
      acc[t] = wmma_bf16(a, loadB(bp[t], kc * 32, half), acc[t]);
  }
#pragma unroll
  for (int t = 0; t < FGT; ++t) {
#pragma unroll
    for (int r = 0; r < 8; ++r) {
      const int m = r + half * 8;
      out[(mt * 16 + m) * HIDh + (og * FGT + t) * 16 + nl] = acc[t][r];
    }
  }
}

// ---------------------------------------------------------------------------
extern "C" void kernel_launch(void* const* d_in, const int* in_sizes, int n_in,
                              void* d_out, int out_size, void* d_ws, size_t ws_size,
                              hipStream_t stream) {
  (void)in_sizes; (void)n_in; (void)out_size; (void)ws_size;
  const float* hidden     = (const float*)d_in[0];
  const float* q          = (const float*)d_in[1];
  const float* k          = (const float*)d_in[2];
  const float* v          = (const float*)d_in[3];
  const float* memkv      = (const float*)d_in[4];
  const float* kns        = (const float*)d_in[5];
  const float* vns        = (const float*)d_in[6];
  const float* combiner_w = (const float*)d_in[7];
  const float* combiner_b = (const float*)d_in[8];
  const float* combine_w  = (const float*)d_in[9];
  float* out = (float*)d_out;

  // Workspace carve-up (byte offsets, 256B-aligned chunks)
  char* p = (char*)d_ws;
  auto take = [&p](size_t bytes) { char* r = p; p += (bytes + 255) & ~(size_t)255; return r; };
  __bf16* ckb   = (__bf16*)take((size_t)HKVh * CKROWS * Dd * 2);   // 72 KB
  __bf16* cvT   = (__bf16*)take((size_t)HKVh * Dd * CVS * 2);      // 80 KB
  __bf16* kb    = (__bf16*)take((size_t)HKVh * Nn * Dd * 2);       // 1 MB
  __bf16* vT    = (__bf16*)take((size_t)HKVh * Dd * Nn * 2);       // 1 MB
  __bf16* hb    = (__bf16*)take((size_t)Nn * HIDh * 2);            // 4 MB
  __bf16* cwb   = (__bf16*)take((size_t)3 * Hh * HIDh * 2);        // 96 KB
  __bf16* wb    = (__bf16*)take((size_t)HIDh * HIDh * 2);          // 2 MB
  __bf16* comb  = (__bf16*)take((size_t)Nn * HIDh * 2);            // 4 MB
  float*  imp   = (float*)take((size_t)HKVh * Nn * Ww * 4);        // 4 MB
  int*    sel   = (int*)take((size_t)HKVh * Nn * 4 * 4);           // 128 KB
  float*  oc    = (float*)take((size_t)Hh * Nn * Dd * 4);          // 8 MB
  float*  of_   = (float*)take((size_t)Hh * Nn * Dd * 4);          // 8 MB
  float*  os_   = (float*)take((size_t)Hh * Nn * Dd * 4);          // 8 MB
  float*  gates = (float*)take((size_t)Nn * 3 * Hh * 4);           // 384 KB

  hipMemsetAsync(imp, 0, (size_t)HKVh * Nn * Ww * 4, stream);
  hipMemsetAsync(ckb, 0, (size_t)HKVh * CKROWS * Dd * 2, stream);  // pad rows
  hipMemsetAsync(cvT, 0, (size_t)HKVh * Dd * CVS * 2, stream);     // pad cols

  // bf16 staging
  cvt_bf16_kernel<<<(HKVh * Nn * Dd + 255) / 256, 256, 0, stream>>>(k, kb, HKVh * Nn * Dd);
  transpose_v_kernel<<<(HKVh * Nn * Dd + 255) / 256, 256, 0, stream>>>(v, vT);
  cvt_bf16_kernel<<<(Nn * HIDh + 255) / 256, 256, 0, stream>>>(hidden, hb, Nn * HIDh);
  cvt_bf16_kernel<<<(3 * Hh * HIDh + 255) / 256, 256, 0, stream>>>(combiner_w, cwb, 3 * Hh * HIDh);
  cvt_bf16_kernel<<<(HIDh * HIDh + 255) / 256, 256, 0, stream>>>(combine_w, wb, HIDh * HIDh);
  memkv_fill_kernel<<<HKVh, Dd, 0, stream>>>(memkv, ckb, cvT);

  compress_kernel<<<HKVh * Ww, Dd, 0, stream>>>(k, v, kns, vns, ckb, cvT);
  cmp_attn_kernel<<<Hh * QT, 32, 0, stream>>>(q, ckb, cvT, oc, imp);
  topk_kernel<<<(HKVh * Nn + 255) / 256, 256, 0, stream>>>(imp, sel);
  fine_attn_kernel<<<Hh * Nn, 64, 0, stream>>>(q, k, v, sel, of_);
  win_attn_kernel<<<Hh * QT, 32, 0, stream>>>(q, kb, vT, os_);
  gates_kernel<<<Nn / 16, 32, 0, stream>>>(hb, cwb, combiner_b, gates);
  combine_kernel<<<(Nn * HIDh + 255) / 256, 256, 0, stream>>>(gates, oc, of_, os_, comb);
  final_gemm_kernel<<<(Nn / 16) * (HIDh / (16 * FGT)), 32, 0, stream>>>(comb, wb, out);
}
